// NPUAcceleratedAttention_56642028699648
// MI455X (gfx1250) — matrix operand order repaired and checked by reading, not compiled, not run on hardware
//
#include <hip/hip_runtime.h>
#include <math.h>

// ---------------- CDNA5 WMMA types ----------------
typedef __attribute__((ext_vector_type(16))) __bf16 v16bf;
typedef __attribute__((ext_vector_type(8)))  float  v8f;
typedef int v4i_t __attribute__((vector_size(16)));   // matches async-LDS builtin

struct U4 { unsigned int x, y, z, w; };           // 16-byte POD load unit
union  AB { U4 q[2]; v16bf v; };                  // one WMMA A/B operand (8 VGPRs)

#if defined(__has_builtin)
#if __has_builtin(__builtin_amdgcn_global_load_async_to_lds_b128) && \
    __has_builtin(__builtin_amdgcn_s_wait_asynccnt)
#define USE_ASYNC 1
#endif
#endif
#ifndef USE_ASYNC
#define USE_ASYNC 0
#endif

#define AS1 __attribute__((address_space(1)))
#define AS3 __attribute__((address_space(3)))

#if USE_ASYNC
#define ASY128(g, l) __builtin_amdgcn_global_load_async_to_lds_b128(            \
        (AS1 v4i_t*)(g), (AS3 v4i_t*)(l), 0, 0)
#define WAITB(n) __builtin_amdgcn_s_wait_asynccnt(n)
#else
#define ASY128(g, l) (*(U4*)(l) = *(const U4*)(g))
#define WAITB(n) asm volatile("s_wait_dscnt 0x0" ::: "memory")
#endif

// Raw workgroup split barrier: same sequence the compiler emits for
// __syncthreads(), minus the implicit loadcnt/storecnt flush (ordering of the
// LDS staging is handled explicitly via s_wait_asynccnt).
#define BAR() asm volatile("s_barrier_signal -1\n\ts_barrier_wait -1" ::: "memory")

static __device__ __forceinline__ unsigned short f2bf(float f) {
    unsigned int u = __float_as_uint(f);
    unsigned int r = u + 0x7FFFu + ((u >> 16) & 1u);   // round-to-nearest-even
    return (unsigned short)(r >> 16);
}

static __device__ __forceinline__ v8f wmma_bf16(const AB& a, const AB& b, v8f c) {
    return __builtin_amdgcn_wmma_f32_16x16x32_bf16(
        /*neg_a=*/false, a.v, /*neg_b=*/false, b.v,
        /*c_mod=*/(short)0, c, /*reuse_a=*/false, /*reuse_b=*/false);
}

// ---------------- problem constants ----------------
#define BD 4
#define SD 1500
#define SP 1504          // padded tokens per batch (94 tiles of 16)
#define DD 1280
#define HD 20
#define DH 64

// ---------------- fp32 -> bf16 pack ----------------
__global__ void pack_bf16(const float* __restrict__ src,
                          unsigned short* __restrict__ dst, int n) {
    int i = blockIdx.x * blockDim.x + threadIdx.x;
    if (i < n) dst[i] = f2bf(src[i]);
}

// ---------------- C = A @ W^T (+bias), bf16 WMMA, fp32 acc ----------------
// Block: 256 threads = 8 waves, computes 128 rows x 64 cols (1 M-tile/wave,
// ~70 live VGPRs per wave -> no spills). A and W tiles staged block-wide into
// LDS with async global->LDS, double-buffered with a 2-barrier pipeline.
// mode 0: store bf16 to [4][1504][1280] (q/k, padded-token layout), *scale
// mode 1: store bf16 transposed per head -> vT [4][20][64][1504]
// mode 2: store fp32 row-major [6000][1280] to d_out
__global__ void gemm_bt(const unsigned short* __restrict__ A,
                        const unsigned short* __restrict__ W,
                        const float* __restrict__ bias,
                        unsigned short* __restrict__ outb,
                        float* __restrict__ outf,
                        int mode, float scale)
{
    __shared__ __align__(16) unsigned short ldsA[2][128][32]; // [buf][row][k] 16KB
    __shared__ __align__(16) unsigned short ldsB[2][64][32];  // [buf][n][k]    8KB
    const int tid  = threadIdx.x;
    const int lane = tid & 31;
    const int wave = tid >> 5;                         // 0..7
    const int l16  = lane & 15;
    const int half = lane >> 4;
    const int n0   = blockIdx.y * 64;
    const int mtile = blockIdx.x * 8 + wave;           // 0..375 (375 skipped)

    // A stage: 128 rows x 64B per K-step; 2 threads per row, 32B each.
    int arow_g = blockIdx.x * 128 + (tid >> 1);
    if (arow_g > 5999) arow_g = 5999;                  // dup rows feed only the
    const int aco = (tid & 1) * 16;                    // skipped tile 375
    const unsigned short* asrc = A + (size_t)arow_g * DD + aco;
    // B stage: 64 rows x 64B (= W[n0+row][kk..kk+31]); 4 threads per row.
    const int brow = tid >> 2;
    const int bco  = (tid & 3) * 8;
    const unsigned short* wsrc = W + (size_t)(n0 + brow) * DD + bco;

#define STAGE(it, buf)                                                         \
    do {                                                                       \
        const int kk_ = (it) * 32;                                             \
        ASY128(asrc + kk_,     &ldsA[buf][tid >> 1][aco]);                     \
        ASY128(asrc + kk_ + 8, &ldsA[buf][tid >> 1][aco + 8]);                 \
        ASY128(wsrc + kk_,     &ldsB[buf][brow][bco]);                         \
    } while (0)

    v8f acc[4] = {};
#define COMPUTE(buf)                                                           \
    do {                                                                       \
        AB at;                                                                 \
        at.q[0] = *(const U4*)&ldsA[buf][wave * 16 + l16][half * 8];           \
        at.q[1] = *(const U4*)&ldsA[buf][wave * 16 + l16][16 + half * 8];      \
        _Pragma("unroll")                                                      \
        for (int nt = 0; nt < 4; ++nt) {                                       \
            AB bt;                                                             \
            bt.q[0] = *(const U4*)&ldsB[buf][nt * 16 + l16][half * 8];         \
            bt.q[1] = *(const U4*)&ldsB[buf][nt * 16 + l16][16 + half * 8];    \
            acc[nt] = wmma_bf16(at, bt, acc[nt]);                              \
        }                                                                      \
    } while (0)

    STAGE(0, 0);
    for (int itp = 0; itp < 20; ++itp) {               // 40 K-steps, 2 per trip
        BAR();                                         // buf1 readers done
        STAGE(2 * itp + 1, 1);
        WAITB(3);                                      // stage 2*itp complete
        BAR();                                         // visible block-wide
        COMPUTE(0);
        BAR();                                         // buf0 readers done
        if (itp < 19) {
            STAGE(2 * itp + 2, 0);
            WAITB(3);
        } else {
            WAITB(0);
        }
        BAR();
        COMPUTE(1);
    }

    if (mtile < 375) {
#pragma unroll
        for (int nt = 0; nt < 4; ++nt) {
#pragma unroll
            for (int r = 0; r < 8; ++r) {
                const int grow = mtile * 16 + r + 8 * half;   // < 6000
                const int col  = n0 + nt * 16 + l16;
                float v = acc[nt][r];
                if (bias) v += bias[col];
                v *= scale;
                if (mode == 0) {
                    const int bb = grow / SD, ss = grow % SD;
                    outb[(size_t)(bb * SP + ss) * DD + col] = f2bf(v);
                } else if (mode == 1) {
                    const int bb = grow / SD, ss = grow % SD;
                    const int h = col >> 6, dd = col & 63;
                    outb[(size_t)((bb * HD + h) * DH + dd) * SP + ss] = f2bf(v);
                } else {
                    outf[(size_t)grow * DD + col] = v;
                }
            }
        }
    }
}

// ---------------- flash attention ----------------
// Block: 128 threads = 4 waves, all sharing one (b,h); wave w handles query
// block blockIdx.x*4+w (clamped to 93 -> benign duplicate compute/stores).
// K (32x64) and V (64x32, from vT) tiles for each 32-key step are staged
// block-wide via async global->LDS, double-buffered; shared by all 4 waves.
__global__ void flash_attn(const unsigned short* __restrict__ qb,
                           const unsigned short* __restrict__ kb,
                           const unsigned short* __restrict__ vT,
                           unsigned short* __restrict__ ctxb)
{
    __shared__ __align__(16) unsigned short ldsK[2][32][64];  // [buf][key][dim] 8KB
    __shared__ __align__(16) unsigned short ldsV[2][64][32];  // [buf][dim][key] 8KB
    __shared__ __align__(16) unsigned short ldsP[4][16][32];  // per-wave P tile 4KB
    const int tid  = threadIdx.x;
    const int lane = tid & 31;
    const int wave = tid >> 5;
    int qblk = blockIdx.x * 4 + wave;
    if (qblk > 93) qblk = 93;            // keep wave alive for staging/barriers
    const int h    = blockIdx.y;
    const int b    = blockIdx.z;
    const int l16  = lane & 15;
    const int half = lane >> 4;

    // q A-tiles for K(dh)=0..31 and 32..63, loaded once
    const unsigned short* qrow =
        qb + (size_t)(b * SP + qblk * 16 + l16) * DD + h * DH;
    AB aq0, aq1;
    aq0.q[0] = *(const U4*)(qrow + half * 8);
    aq0.q[1] = *(const U4*)(qrow + 16 + half * 8);
    aq1.q[0] = *(const U4*)(qrow + 32 + half * 8);
    aq1.q[1] = *(const U4*)(qrow + 48 + half * 8);

    v8f acc[4] = {};
    float mrun[8], lrun[8];
#pragma unroll
    for (int r = 0; r < 8; ++r) { mrun[r] = -INFINITY; lrun[r] = 0.f; }

    const unsigned short* kbase = kb + (size_t)b * SP * DD + h * DH;
    const unsigned short* vbase = vT + (size_t)((b * HD + h) * DH) * SP;

    // stage mapping: K tile 32 rows x 128B (4 thr/row), V tile 64 rows x 64B
    const int krow = tid >> 2, kco = (tid & 3) * 16;
    const int vrow = tid >> 1, vco = (tid & 1) * 16;

#define STAGEKV(it, buf)                                                       \
    do {                                                                       \
        const int kk_ = (it) * 32;                                             \
        const unsigned short* gk = kbase + (size_t)(kk_ + krow) * DD + kco;    \
        ASY128(gk,     &ldsK[buf][krow][kco]);                                 \
        ASY128(gk + 8, &ldsK[buf][krow][kco + 8]);                             \
        const unsigned short* gv = vbase + (size_t)vrow * SP + kk_ + vco;      \
        ASY128(gv,     &ldsV[buf][vrow][vco]);                                 \
        ASY128(gv + 8, &ldsV[buf][vrow][vco + 8]);                             \
    } while (0)

    // one 32-key step: scores (4 WMMA) -> online softmax -> P@V (4 WMMA)
#define PROCESS(k0v, buf)                                                      \
    do {                                                                       \
        const int k0_ = (k0v);                                                 \
        v8f s0 = {}, s1 = {};                                                  \
        AB kt;                                                                 \
        kt.q[0] = *(const U4*)&ldsK[buf][l16][half * 8];                       \
        kt.q[1] = *(const U4*)&ldsK[buf][l16][16 + half * 8];                  \
        s0 = wmma_bf16(aq0, kt, s0);                                           \
        kt.q[0] = *(const U4*)&ldsK[buf][l16][32 + half * 8];                  \
        kt.q[1] = *(const U4*)&ldsK[buf][l16][48 + half * 8];                  \
        s0 = wmma_bf16(aq1, kt, s0);                                           \
        kt.q[0] = *(const U4*)&ldsK[buf][16 + l16][half * 8];                  \
        kt.q[1] = *(const U4*)&ldsK[buf][16 + l16][16 + half * 8];             \
        s1 = wmma_bf16(aq0, kt, s1);                                           \
        kt.q[0] = *(const U4*)&ldsK[buf][16 + l16][32 + half * 8];             \
        kt.q[1] = *(const U4*)&ldsK[buf][16 + l16][48 + half * 8];             \
        s1 = wmma_bf16(aq1, kt, s1);                                           \
        if (k0_ + l16 >= SD) {                                                 \
            _Pragma("unroll") for (int r = 0; r < 8; ++r) s0[r] = -INFINITY;   \
        }                                                                      \
        if (k0_ + 16 + l16 >= SD) {                                            \
            _Pragma("unroll") for (int r = 0; r < 8; ++r) s1[r] = -INFINITY;   \
        }                                                                      \
        _Pragma("unroll")                                                      \
        for (int r = 0; r < 8; ++r) {                                          \
            float mx = fmaxf(s0[r], s1[r]);                                    \
            mx = fmaxf(mx, __shfl_xor(mx, 1));                                 \
            mx = fmaxf(mx, __shfl_xor(mx, 2));                                 \
            mx = fmaxf(mx, __shfl_xor(mx, 4));                                 \
            mx = fmaxf(mx, __shfl_xor(mx, 8));                                 \
            const float mn    = fmaxf(mrun[r], mx);                            \
            const float alpha = __expf(mrun[r] - mn);                          \
            mrun[r] = mn;                                                      \
            lrun[r] *= alpha;                                                  \
            acc[0][r] *= alpha; acc[1][r] *= alpha;                            \
            acc[2][r] *= alpha; acc[3][r] *= alpha;                            \
            const float p0 = __expf(s0[r] - mn);                               \
            const float p1 = __expf(s1[r] - mn);                               \
            s0[r] = p0; s1[r] = p1;                                            \
            float rs = p0 + p1;                                                \
            rs += __shfl_xor(rs, 1);                                           \
            rs += __shfl_xor(rs, 2);                                           \
            rs += __shfl_xor(rs, 4);                                           \
            rs += __shfl_xor(rs, 8);                                           \
            lrun[r] += rs;                                                     \
        }                                                                      \
        _Pragma("unroll")                                                      \
        for (int r = 0; r < 8; ++r) {                                          \
            ldsP[wave][r + 8 * half][l16]      = f2bf(s0[r]);                  \
            ldsP[wave][r + 8 * half][16 + l16] = f2bf(s1[r]);                  \
        }                                                                      \
        asm volatile("" ::: "memory");                                         \
        AB ap;                                                                 \
        ap.q[0] = *(const U4*)&ldsP[wave][l16][half * 8];                      \
        ap.q[1] = *(const U4*)&ldsP[wave][l16][16 + half * 8];                 \
        asm volatile("" ::: "memory");                                         \
        _Pragma("unroll")                                                      \
        for (int nt = 0; nt < 4; ++nt) {                                       \
            AB vt;                                                             \
            vt.q[0] = *(const U4*)&ldsV[buf][nt * 16 + l16][half * 8];         \
            vt.q[1] = *(const U4*)&ldsV[buf][nt * 16 + l16][16 + half * 8];    \
            acc[nt] = wmma_bf16(ap, vt, acc[nt]);                              \
        }                                                                      \
    } while (0)

    STAGEKV(0, 0);
    for (int p = 0; p < 23; ++p) {          // steps 0..45; step 46 peeled below
        BAR();                              // buf1 readers done
        STAGEKV(2 * p + 1, 1);
        WAITB(4);                           // stage 2p complete
        BAR();
        PROCESS((2 * p) * 32, 0);
        BAR();                              // buf0 readers done
        STAGEKV(2 * p + 2, 0);              // p=22 -> stage 46 (keys 1472..1503)
        WAITB(4);
        BAR();
        PROCESS((2 * p + 1) * 32, 1);
    }
    BAR();
    WAITB(0);
    BAR();
    PROCESS(46 * 32, 0);

    // ---- normalize and store context (skip padded query rows)
#pragma unroll
    for (int r = 0; r < 8; ++r) {
        const float inv = 1.0f / lrun[r];
        const int srow = qblk * 16 + r + 8 * half;
        if (srow < SD) {
            const size_t base = (size_t)(b * SD + srow) * DD + h * DH;
#pragma unroll
            for (int nt = 0; nt < 4; ++nt)
                ctxb[base + nt * 16 + l16] = f2bf(acc[nt][r] * inv);
        }
    }
}

// ---------------- host launch ----------------
extern "C" void kernel_launch(void* const* d_in, const int* in_sizes, int n_in,
                              void* d_out, int out_size, void* d_ws, size_t ws_size,
                              hipStream_t stream) {
    (void)in_sizes; (void)n_in; (void)out_size; (void)ws_size;
    const float* x  = (const float*)d_in[0];
    const float* Wq = (const float*)d_in[1];
    const float* bq = (const float*)d_in[2];
    const float* Wk = (const float*)d_in[3];
    const float* Wv = (const float*)d_in[4];
    const float* bv = (const float*)d_in[5];
    const float* Wo = (const float*)d_in[6];
    const float* bo = (const float*)d_in[7];
    float* out = (float*)d_out;

    char* ws = (char*)d_ws;
    size_t off = 0;
    auto alloc = [&](size_t bytes) -> void* {
        void* p = ws + off;
        off = (off + bytes + 255) & ~(size_t)255;
        return p;
    };
    const size_t nX = (size_t)BD * SD * DD;        // 7,680,000
    const size_t nW = (size_t)DD * DD;             // 1,638,400
    unsigned short* xb   = (unsigned short*)alloc(nX * 2);
    unsigned short* wqb  = (unsigned short*)alloc(nW * 2);
    unsigned short* wkb  = (unsigned short*)alloc(nW * 2);
    unsigned short* wvb  = (unsigned short*)alloc(nW * 2);
    unsigned short* wob  = (unsigned short*)alloc(nW * 2);
    unsigned short* qbuf = (unsigned short*)alloc((size_t)BD * SP * DD * 2);
    unsigned short* kbuf = (unsigned short*)alloc((size_t)BD * SP * DD * 2);
    unsigned short* vT   = (unsigned short*)alloc((size_t)BD * HD * DH * SP * 2);
    unsigned short* ctxb = (unsigned short*)alloc(nX * 2);

    pack_bf16<<<(int)((nX + 255) / 256), 256, 0, stream>>>(x,  xb,  (int)nX);
    pack_bf16<<<(int)((nW + 255) / 256), 256, 0, stream>>>(Wq, wqb, (int)nW);
    pack_bf16<<<(int)((nW + 255) / 256), 256, 0, stream>>>(Wk, wkb, (int)nW);
    pack_bf16<<<(int)((nW + 255) / 256), 256, 0, stream>>>(Wv, wvb, (int)nW);
    pack_bf16<<<(int)((nW + 255) / 256), 256, 0, stream>>>(Wo, wob, (int)nW);

    dim3 gg(47, 20, 1);   // blocks of 8 waves: 128 rows x 64 cols each
    gemm_bt<<<gg, 256, 0, stream>>>(xb, wqb, bq,      qbuf, nullptr, 0, 0.125f);
    gemm_bt<<<gg, 256, 0, stream>>>(xb, wkb, nullptr, kbuf, nullptr, 0, 1.0f);
    gemm_bt<<<gg, 256, 0, stream>>>(xb, wvb, bv,      vT,   nullptr, 1, 1.0f);

    dim3 fg(24, 20, 4);   // 94 query blocks / 4 waves, H, B
    flash_attn<<<fg, 128, 0, stream>>>(qbuf, kbuf, vT, ctxb);

    gemm_bt<<<gg, 256, 0, stream>>>(ctxb, wob, bo, nullptr, out, 2, 1.0f);
}
